// IFOffsetWarpAttn_82025285419165
// MI455X (gfx1250) — matrix operand order repaired and checked by reading, not compiled, not run on hardware
//
#include <hip/hip_runtime.h>
#include <hip/hip_bf16.h>

#define NB 2
#define CC 64
#define HH 96
#define WW 96
#define NPIX (NB*HH*WW)

typedef __attribute__((ext_vector_type(16))) __bf16 v16bf;
typedef __attribute__((ext_vector_type(8)))  float  v8f;

struct __align__(16) U4 { unsigned int x, y, z, w; };
union AF { U4 q[2]; v16bf bf; };

__device__ __forceinline__ unsigned short f2bf(float f) {
    unsigned int u = __float_as_uint(f);
    unsigned int r = (u + 0x7FFFu + ((u >> 16) & 1u)) >> 16;
    return (unsigned short)r;
}

constexpr int pickGCH(int n) {
    for (int g = 8; g > 1; --g)
        if (n % g == 0) return g;
    return 1;
}

// ---------------------------------------------------------------------------
// Weight swizzle: OIHW f32 -> bf16 fragment-ordered B matrix
// dst[((q*4+nt)*32 + lane)*16 + e] = W[n=nt*16+(lane&15)][k = q*32 + (lane>=16?16:0) + e]
// with K index k = tap*Cin + ci, tap = ky*KW+kx
// ---------------------------------------------------------------------------
__global__ void swizzle_w_kernel(const float* __restrict__ W, int Cin, int KH, int KW,
                                 unsigned short* __restrict__ dst, int Ktot) {
    int idx = blockIdx.x * blockDim.x + threadIdx.x;
    if (idx >= Ktot * 64) return;
    int e  = idx & 15;
    int l  = (idx >> 4) & 31;
    int nt = (idx >> 9) & 3;
    int q  = idx >> 11;
    int n  = nt * 16 + (l & 15);
    int k  = q * 32 + ((l >> 4) << 4) + e;
    int tap = k / Cin;
    int ci  = k - tap * Cin;
    int ky  = tap / KW, kx = tap - ky * KW;
    dst[idx] = f2bf(W[((size_t)(n * Cin + ci) * KH + ky) * KW + kx]);
}

// ---------------------------------------------------------------------------
// Offset net conv1: hidden = relu(conv3x3(concat[key,sup,key-sup], w_off1))
// ---------------------------------------------------------------------------
__global__ void offset_hidden_kernel(const float* __restrict__ sup, const float* __restrict__ key,
                                     const float* __restrict__ w1, float* __restrict__ hidden) {
    int idx = blockIdx.x * blockDim.x + threadIdx.x;
    if (idx >= NB * 32 * HH * WW) return;
    int x = idx % WW, y = (idx / WW) % HH, oc = (idx / (WW * HH)) % 32, b = idx / (WW * HH * 32);
    float acc = 0.f;
    for (int ci = 0; ci < CC; ++ci) {
        const float* kb = key + (size_t)(b * CC + ci) * HH * WW;
        const float* sb = sup + (size_t)(b * CC + ci) * HH * WW;
        const float* wk = w1 + (size_t)(oc * 192 + ci) * 9;
        const float* ws = w1 + (size_t)(oc * 192 + ci + 64) * 9;
        const float* wd = w1 + (size_t)(oc * 192 + ci + 128) * 9;
        #pragma unroll
        for (int t = 0; t < 9; ++t) {
            int yy = y + t / 3 - 1, xx = x + t % 3 - 1;
            if (yy < 0 || yy >= HH || xx < 0 || xx >= WW) continue;
            float kv = kb[yy * WW + xx], sv = sb[yy * WW + xx];
            acc += wk[t] * kv + ws[t] * sv + wd[t] * (kv - sv);
        }
    }
    hidden[idx] = fmaxf(acc, 0.f);
}

// ---------------------------------------------------------------------------
// Offset net conv2 -> sample coordinates (pixel units, border-clipped)
// ---------------------------------------------------------------------------
__global__ void offset_flow_kernel(const float* __restrict__ hidden, const float* __restrict__ w2,
                                   float* __restrict__ grid) {
    int idx = blockIdx.x * blockDim.x + threadIdx.x;
    if (idx >= NPIX) return;
    int x = idx % WW, y = (idx / WW) % HH, b = idx / (WW * HH);
    float o0 = 0.f, o1 = 0.f;
    for (int hc = 0; hc < 32; ++hc) {
        const float* hb  = hidden + (size_t)(b * 32 + hc) * HH * WW;
        const float* w0p = w2 + (size_t)hc * 9;
        const float* w1p = w2 + (size_t)(32 + hc) * 9;
        #pragma unroll
        for (int t = 0; t < 9; ++t) {
            int yy = y + t / 3 - 1, xx = x + t % 3 - 1;
            if (yy < 0 || yy >= HH || xx < 0 || xx >= WW) continue;
            float hv = hb[yy * WW + xx];
            o0 += w0p[t] * hv;
            o1 += w1p[t] * hv;
        }
    }
    grid[(size_t)idx * 2 + 0] = fminf(fmaxf((float)x + o0, 0.f), (float)(WW - 1));
    grid[(size_t)idx * 2 + 1] = fminf(fmaxf((float)y + o1, 0.f), (float)(HH - 1));
}

// ---------------------------------------------------------------------------
// Bilinear warp (border) + transposes: warpT f32 BHWC, x_bf16 ch 0..63, keyT f32 BHWC
// ---------------------------------------------------------------------------
__global__ void warp_kernel(const float* __restrict__ sup, const float* __restrict__ key,
                            const float* __restrict__ grid, float* __restrict__ warpT,
                            float* __restrict__ keyT, unsigned short* __restrict__ xbf) {
    int idx = blockIdx.x * blockDim.x + threadIdx.x;
    if (idx >= NPIX) return;
    int x = idx % WW, y = (idx / WW) % HH, b = idx / (WW * HH);
    float px = grid[(size_t)idx * 2], py = grid[(size_t)idx * 2 + 1];
    float x0f = floorf(px), y0f = floorf(py);
    int x0 = (int)x0f; x0 = x0 < 0 ? 0 : (x0 > WW - 1 ? WW - 1 : x0);
    int y0 = (int)y0f; y0 = y0 < 0 ? 0 : (y0 > HH - 1 ? HH - 1 : y0);
    int x1 = x0 + 1 > WW - 1 ? WW - 1 : x0 + 1;
    int y1 = y0 + 1 > HH - 1 ? HH - 1 : y0 + 1;
    float wx = px - x0f, wy = py - y0f;
    for (int c = 0; c < CC; ++c) {
        const float* sb = sup + (size_t)(b * CC + c) * HH * WW;
        float g00 = sb[y0 * WW + x0], g01 = sb[y0 * WW + x1];
        float g10 = sb[y1 * WW + x0], g11 = sb[y1 * WW + x1];
        float v = (1.f - wy) * ((1.f - wx) * g00 + wx * g01) + wy * ((1.f - wx) * g10 + wx * g11);
        warpT[(size_t)idx * CC + c] = v;
        xbf[(size_t)idx * 256 + c] = f2bf(v);
        keyT[(size_t)idx * CC + c] = key[(size_t)(b * CC + c) * HH * WW + y * WW + x];
    }
}

// ---------------------------------------------------------------------------
// Implicit-GEMM conv with WMMA bf16 (16x16x32).
// One wave = 16 pixels x 64 outputs; block = 8 waves sharing weights via LDS
// (double-buffered, one barrier per K-group). Grid is exact: no early return.
// OUTMODE: 0 = f32 BHWC (+opt bias/bn), 1 = bf16 BHWC (bias+bn+relu),
//          2 = f32 NCHW (+bn+shortcut+relu)
// ---------------------------------------------------------------------------
template <int ROWSTRIDE, int CIN, int TAPS, int OUTMODE>
__global__ void __launch_bounds__(256) gemm_conv_kernel(
    const unsigned short* __restrict__ A, const unsigned short* __restrict__ Wsw,
    const float* __restrict__ bias, const float* __restrict__ bng,
    const float* __restrict__ bnb, const float* __restrict__ addIn,
    float* __restrict__ outF, unsigned short* __restrict__ outH) {
    constexpr int NCH  = (CIN * TAPS) / 32;   // total K chunks
    constexpr int CPT  = CIN / 32;            // chunks per tap
    constexpr int GCH  = pickGCH(NCH);        // chunks per LDS group (<=8)
    constexpr int NGRP = NCH / GCH;

    __shared__ U4 wlds[2 * GCH * 256];        // double-buffered weights (<=64KB)

    const int tilesPerRow = WW / 16;          // 6
    int tid  = threadIdx.x;
    int wid  = (blockIdx.x * blockDim.x + tid) >> 5;
    int lane = tid & 31;
    int x0 = (wid % tilesPerRow) * 16;
    int y  = (wid / tilesPerRow) % HH;
    int b  =  wid / (tilesPerRow * HH);
    int m   = lane & 15;
    int sub = (lane >> 4) << 4;

    v8f accs[4] = {{}, {}, {}, {}};

    // preload group 0
    {
        const U4* src = (const U4*)Wsw;
        #pragma unroll
        for (int i = 0; i < GCH; ++i) wlds[tid + i * 256] = src[tid + i * 256];
    }
    __syncthreads();

    for (int g = 0; g < NGRP; ++g) {
        if (g + 1 < NGRP) {   // prefetch next weight group into other buffer
            const U4* src = (const U4*)Wsw + (size_t)(g + 1) * GCH * 256;
            U4* dst = wlds + ((g + 1) & 1) * GCH * 256;
            #pragma unroll
            for (int i = 0; i < GCH; ++i) dst[tid + i * 256] = src[tid + i * 256];
        }
        const unsigned short* wb = (const unsigned short*)(wlds + (g & 1) * GCH * 256);
        #pragma unroll
        for (int qq = 0; qq < GCH; ++qq) {
            int q = g * GCH + qq;
            int tap  = q / CPT;               // CPT is constexpr -> shifts
            int cOff = (q - tap * CPT) << 5;
            int dy = (TAPS == 9) ? (tap / 3 - 1) : 0;
            int dx = (TAPS == 9) ? (tap % 3 - 1) : 0;
            int yy = y + dy;
            int xx = x0 + m + dx;
            AF a;
            if (yy >= 0 && yy < HH && xx >= 0 && xx < WW) {
                const unsigned short* ap =
                    A + (size_t)((b * HH + yy) * WW + xx) * ROWSTRIDE + cOff + sub;
                a.q[0] = *(const U4*)(ap);
                a.q[1] = *(const U4*)(ap + 8);
            } else {
                a.q[0] = U4{0, 0, 0, 0};
                a.q[1] = U4{0, 0, 0, 0};
            }
            #pragma unroll
            for (int nt = 0; nt < 4; ++nt) {
                AF bm;
                const unsigned short* wp = wb + ((qq * 4 + nt) * 32 + lane) * 16;
                bm.q[0] = *(const U4*)(wp);
                bm.q[1] = *(const U4*)(wp + 8);
                accs[nt] = __builtin_amdgcn_wmma_f32_16x16x32_bf16(
                    false, a.bf, false, bm.bf, (short)0, accs[nt], false, false);
            }
        }
        __syncthreads();
    }

    const float inv = 0.9999950000374997f;  // 1/sqrt(1+1e-5)
    int col = lane & 15;
    int rb  = (lane >> 4) << 3;
    #pragma unroll
    for (int nt = 0; nt < 4; ++nt) {
        int ch = nt * 16 + col;
        float g   = bng ? bng[ch] : 1.f;
        float bb2 = bnb ? bnb[ch] : 0.f;
        float bs  = bias ? bias[ch] : 0.f;
        #pragma unroll
        for (int r = 0; r < 8; ++r) {
            int mm  = rb + r;
            int pix = (b * HH + y) * WW + x0 + mm;
            float v = accs[nt][r] + bs;
            if (bng) v = v * inv * g + bb2;
            if constexpr (OUTMODE == 2) {
                v += addIn[(size_t)pix * 64 + ch];
                v = fmaxf(v, 0.f);
                outF[((size_t)(b * 64 + ch) * HH + y) * WW + x0 + mm] = v;
            } else if constexpr (OUTMODE == 1) {
                outH[(size_t)pix * 64 + ch] = f2bf(fmaxf(v, 0.f));
            } else {
                outF[(size_t)pix * 64 + ch] = v;
            }
        }
    }
}

// ---------------------------------------------------------------------------
// Neighborhood attention, one thread per pixel, writes bf16 into x concat slot
// ---------------------------------------------------------------------------
template <int K>
__global__ void natten_kernel(const float* __restrict__ keyT, const float* __restrict__ warpT,
                              const float* __restrict__ V, const float* __restrict__ rpb,
                              unsigned short* __restrict__ xbf, int chanBase) {
    int idx = blockIdx.x * blockDim.x + threadIdx.x;
    if (idx >= NPIX) return;
    int x = idx % WW, y = (idx / WW) % HH, b = idx / (WW * HH);
    const int R = K / 2;
    int si = y - R; si = si < 0 ? 0 : (si > HH - K ? HH - K : si);
    int sj = x - R; sj = sj < 0 ? 0 : (sj > WW - K ? WW - K : sj);
    const float* qp = keyT + (size_t)idx * CC;
    float s[K * K];
    float mx = -1e30f;
    #pragma unroll
    for (int ii = 0; ii < K; ++ii) {
        #pragma unroll
        for (int jj = 0; jj < K; ++jj) {
            int ky = si + ii, kx = sj + jj;
            const float* kp = warpT + (size_t)((b * HH + ky) * WW + kx) * CC;
            float d = 0.f;
            for (int c = 0; c < CC; ++c) d += qp[c] * kp[c];
            d += rpb[(ky - y + K - 1) * (2 * K - 1) + (kx - x + K - 1)];
            s[ii * K + jj] = d;
            mx = fmaxf(mx, d);
        }
    }
    float den = 0.f;
    #pragma unroll
    for (int n = 0; n < K * K; ++n) { s[n] = __expf(s[n] - mx); den += s[n]; }
    float rd = 1.f / den;
    #pragma unroll
    for (int n = 0; n < K * K; ++n) s[n] *= rd;
    for (int c = 0; c < CC; ++c) {
        float o = 0.f;
        #pragma unroll
        for (int ii = 0; ii < K; ++ii) {
            #pragma unroll
            for (int jj = 0; jj < K; ++jj)
                o += s[ii * K + jj] * V[(size_t)((b * HH + si + ii) * WW + sj + jj) * CC + c];
        }
        xbf[(size_t)idx * 256 + chanBase + c] = f2bf(o);
    }
}

// ---------------------------------------------------------------------------
extern "C" void kernel_launch(void* const* d_in, const int* in_sizes, int n_in,
                              void* d_out, int out_size, void* d_ws, size_t ws_size,
                              hipStream_t stream) {
    const float* sup   = (const float*)d_in[0];
    const float* key   = (const float*)d_in[1];
    const float* woff1 = (const float*)d_in[2];
    const float* woff2 = (const float*)d_in[3];
    const float* rpb[3] = {(const float*)d_in[4], (const float*)d_in[5], (const float*)d_in[6]};
    const float* wv[3]  = {(const float*)d_in[7], (const float*)d_in[8], (const float*)d_in[9]};
    const float* fw1  = (const float*)d_in[10];
    const float* fb1  = (const float*)d_in[11];
    const float* bn1g = (const float*)d_in[12];
    const float* bn1b = (const float*)d_in[13];
    const float* fw2  = (const float*)d_in[14];
    const float* fb2  = (const float*)d_in[15];
    const float* bn2g = (const float*)d_in[16];
    const float* bn2b = (const float*)d_in[17];
    const float* sw   = (const float*)d_in[18];
    const float* sb   = (const float*)d_in[19];
    const float* sbng = (const float*)d_in[20];
    const float* sbnb = (const float*)d_in[21];
    float* out = (float*)d_out;

    char* ws = (char*)d_ws;
    size_t off = 0;
    auto alloc = [&](size_t bytes) { void* p = ws + off; off = (off + bytes + 255) & ~(size_t)255; return p; };

    float* hidden = (float*)alloc((size_t)NB * 32 * HH * WW * 4);
    float* grid   = (float*)alloc((size_t)NPIX * 2 * 4);
    float* warpT  = (float*)alloc((size_t)NPIX * CC * 4);
    float* keyT   = (float*)alloc((size_t)NPIX * CC * 4);
    unsigned short* xbf = (unsigned short*)alloc((size_t)NPIX * 256 * 2);
    float* Vbuf   = (float*)alloc((size_t)NPIX * CC * 4);
    float* shortcut = (float*)alloc((size_t)NPIX * CC * 4);
    unsigned short* o1 = (unsigned short*)alloc((size_t)NPIX * CC * 2);
    unsigned short* wsw_fw1 = (unsigned short*)alloc((size_t)2304 * 64 * 2);
    unsigned short* wsw_fw2 = (unsigned short*)alloc((size_t)576 * 64 * 2);
    unsigned short* wsw_sw  = (unsigned short*)alloc((size_t)256 * 64 * 2);
    unsigned short* wsw_wv[3];
    for (int l = 0; l < 3; ++l) wsw_wv[l] = (unsigned short*)alloc((size_t)64 * 64 * 2);

    auto blocks = [](int total) { return (total + 255) / 256; };

    // weight swizzles
    swizzle_w_kernel<<<blocks(2304 * 64), 256, 0, stream>>>(fw1, 256, 3, 3, wsw_fw1, 2304);
    swizzle_w_kernel<<<blocks(576 * 64), 256, 0, stream>>>(fw2, 64, 3, 3, wsw_fw2, 576);
    swizzle_w_kernel<<<blocks(256 * 64), 256, 0, stream>>>(sw, 256, 1, 1, wsw_sw, 256);
    for (int l = 0; l < 3; ++l)
        swizzle_w_kernel<<<blocks(64 * 64), 256, 0, stream>>>(wv[l], 64, 1, 1, wsw_wv[l], 64);

    // offset net + warp
    offset_hidden_kernel<<<blocks(NB * 32 * HH * WW), 256, 0, stream>>>(sup, key, woff1, hidden);
    offset_flow_kernel<<<blocks(NPIX), 256, 0, stream>>>(hidden, woff2, grid);
    warp_kernel<<<blocks(NPIX), 256, 0, stream>>>(sup, key, grid, warpT, keyT, xbf);

    // attention levels (stream-ordered, V buffer reused)
    const int gemmBlocks = (NB * HH * (WW / 16)) / 8;  // 1152 waves -> 144 blocks, exact
    for (int l = 0; l < 3; ++l) {
        gemm_conv_kernel<256, 64, 1, 0><<<gemmBlocks, 256, 0, stream>>>(
            xbf, wsw_wv[l], nullptr, nullptr, nullptr, nullptr, Vbuf, nullptr);
        int cb = 64 * (l + 1);
        if (l == 0)
            natten_kernel<3><<<blocks(NPIX), 256, 0, stream>>>(keyT, warpT, Vbuf, rpb[0], xbf, cb);
        else if (l == 1)
            natten_kernel<5><<<blocks(NPIX), 256, 0, stream>>>(keyT, warpT, Vbuf, rpb[1], xbf, cb);
        else
            natten_kernel<7><<<blocks(NPIX), 256, 0, stream>>>(keyT, warpT, Vbuf, rpb[2], xbf, cb);
    }

    // shortcut = bn(conv1x1(x, sw) + sb)
    gemm_conv_kernel<256, 256, 1, 0><<<gemmBlocks, 256, 0, stream>>>(
        xbf, wsw_sw, sb, sbng, sbnb, nullptr, shortcut, nullptr);

    // o1 = relu(bn(conv3x3(x, fw1) + fb1))  (bf16 BHWC)
    gemm_conv_kernel<256, 256, 9, 1><<<gemmBlocks, 256, 0, stream>>>(
        xbf, wsw_fw1, fb1, bn1g, bn1b, nullptr, nullptr, o1);

    // out = relu(bn(conv3x3(o1, fw2) + fb2) + shortcut)  (f32 NCHW)
    gemm_conv_kernel<64, 64, 9, 2><<<gemmBlocks, 256, 0, stream>>>(
        o1, wsw_fw2, fb2, bn2g, bn2b, shortcut, out, nullptr);

    (void)in_sizes; (void)n_in; (void)out_size; (void)ws_size;
}
